// SwinTransformerWithUpsample_8134668059226
// MI455X (gfx1250) — matrix operand order repaired and checked by reading, not compiled, not run on hardware
//
#include <hip/hip_runtime.h>
#include <hip/hip_bf16.h>
#include <math.h>

typedef __attribute__((ext_vector_type(16))) __bf16 bf16x16;
typedef __attribute__((ext_vector_type(8)))  float  floatx8;

#define N_TOK 49
#define C_DIM 192
#define NHEAD 6
#define NPAD  64
#define NWAVE 8

// LDS strides (elements), all rows 16B-aligned
#define XS 196
#define HS 200
#define QS 584
#define VS 72
#define SS 68
#define PS 72
#define AS 200
#define GS 200

// LDS byte offsets
#define OFF_X 0
#define OFF_H 50176
#define OFF_Q 75776
#define OFF_V 150528
#define OFF_S 178176
#define OFF_P 195584
#define OFF_A 204800
#define OFF_G 230400
#define SMEM_BYTES 256000

// workspace byte offsets
#define WS_QKVW 0
#define WS_PROJW 221184
#define WS_FC1W 294912
#define WS_FC2W 589824
#define WS_BIAS 884736

__device__ __forceinline__ floatx8 wmma_bf16(bf16x16 a, bf16x16 b, floatx8 c) {
  return __builtin_amdgcn_wmma_f32_16x16x32_bf16(false, a, false, b, (short)0, c,
                                                 false, false);
}

// A operand: 16x32 (MxK) tile from row-major bf16 buffer.
// lane L (m=L&15, half=L>>4) holds K = {half*8+0..7, 16+half*8+0..7}
__device__ __forceinline__ bf16x16 load_tile_a(const __bf16* base, int r0, int k0,
                                               int stride, int lane) {
  int m = lane & 15, half = lane >> 4;
  const __bf16* p = base + (size_t)(r0 + m) * stride + k0 + half * 8;
  union { bf16x16 v; uint4 q[2]; } t;
  t.q[0] = *reinterpret_cast<const uint4*>(p);
  t.q[1] = *reinterpret_cast<const uint4*>(p + 16);
  return t.v;
}

// B operand: 32x16 (KxN) tile; source stored column-major == row-major [N][K].
// lane L (n=L&15, half=L>>4) holds K = {half*8+0..7, 16+half*8+0..7}
__device__ __forceinline__ bf16x16 load_tile_b(const __bf16* base, int n0, int k0,
                                               int stride, int lane) {
  int n = lane & 15, half = lane >> 4;
  const __bf16* p = base + (size_t)(n0 + n) * stride + k0 + half * 8;
  union { bf16x16 v; uint4 q[2]; } t;
  t.q[0] = *reinterpret_cast<const uint4*>(p);
  t.q[1] = *reinterpret_cast<const uint4*>(p + 16);
  return t.v;
}

__device__ __forceinline__ void layernorm_to_bf16(const float* src, int sstride,
                                                  __bf16* dst, int dstride,
                                                  const float* gamma,
                                                  const float* beta, int w,
                                                  int lane) {
  for (int r = w; r < N_TOK; r += NWAVE) {
    const float* row = src + (size_t)r * sstride;
    float v[6];
    float s = 0.f;
#pragma unroll
    for (int j = 0; j < 6; ++j) { v[j] = row[lane + 32 * j]; s += v[j]; }
#pragma unroll
    for (int off = 16; off > 0; off >>= 1) s += __shfl_xor(s, off, 32);
    float mu = s * (1.f / 192.f);
    float var = 0.f;
#pragma unroll
    for (int j = 0; j < 6; ++j) { float d = v[j] - mu; var += d * d; }
#pragma unroll
    for (int off = 16; off > 0; off >>= 1) var += __shfl_xor(var, off, 32);
    float rinv = rsqrtf(var * (1.f / 192.f) + 1e-5f);
#pragma unroll
    for (int j = 0; j < 6; ++j) {
      int c = lane + 32 * j;
      dst[(size_t)r * dstride + c] = (__bf16)((v[j] - mu) * rinv * gamma[c] + beta[c]);
    }
  }
}

// --- pre-pass: fp32 [R][C] -> bf16 transposed [C][R] (keeps weights in L2) ---
__global__ void transpose_to_bf16(const float* __restrict__ src,
                                  __bf16* __restrict__ dst, int R, int C) {
  int idx = blockIdx.x * blockDim.x + threadIdx.x;
  if (idx < R * C) {
    int r = idx / C, c = idx - r * C;
    dst[(size_t)c * R + r] = (__bf16)src[idx];
  }
}

// --- pre-pass: expand relative position bias table -> [NHEAD][49][49] ---
__global__ void build_rel_bias(const float* __restrict__ table,
                               float* __restrict__ bias) {
  int idx = blockIdx.x * blockDim.x + threadIdx.x;
  if (idx < NHEAD * N_TOK * N_TOK) {
    int h = idx / (N_TOK * N_TOK);
    int rem = idx - h * (N_TOK * N_TOK);
    int i = rem / N_TOK, j = rem - i * N_TOK;
    int yi = i / 7, xi = i % 7, yj = j / 7, xj = j % 7;
    int ridx = (yi - yj + 6) * 13 + (xi - xj + 6);
    bias[idx] = table[ridx * NHEAD + h];
  }
}

// --- fused Swin block: one workgroup (8 wave32) per 7x7 window ---
__global__ __launch_bounds__(256, 1)
void swin_block_kernel(const float* __restrict__ x, const float* __restrict__ ln1_g,
                       const float* __restrict__ ln1_b,
                       const __bf16* __restrict__ qkv_wT,
                       const float* __restrict__ qkv_b,
                       const float* __restrict__ bias_full,
                       const __bf16* __restrict__ proj_wT,
                       const float* __restrict__ proj_b,
                       const float* __restrict__ ln2_g,
                       const float* __restrict__ ln2_b,
                       const __bf16* __restrict__ fc1_wT,
                       const float* __restrict__ fc1_b,
                       const __bf16* __restrict__ fc2_wT,
                       const float* __restrict__ fc2_b,
                       float* __restrict__ out) {
  extern __shared__ char smem[];
  float*  xres = (float*)(smem + OFF_X);   // residual, fp32 [64][196]
  __bf16* hbuf = (__bf16*)(smem + OFF_H);  // LN output   [64][200]
  __bf16* qkvb = (__bf16*)(smem + OFF_Q);  // q|k|v       [64][584]
  __bf16* vT   = (__bf16*)(smem + OFF_V);  // v transposed[192][72]
  float*  Sbuf = (float*)(smem + OFF_S);   // scores      [64][68]
  __bf16* Pbuf = (__bf16*)(smem + OFF_P);  // softmax     [64][72]
  __bf16* aout = (__bf16*)(smem + OFF_A);  // attn out    [64][200]
  __bf16* gbuf = (__bf16*)(smem + OFF_G);  // gelu chunk  [64][200]

  const int tid = threadIdx.x;
  const int lane = tid & 31;
  const int w = tid >> 5;
  const int n = lane & 15;
  const int half = lane >> 4;
  const int mi16 = (w & 3) * 16;  // wave-constant M tile base (all GEMMs)
  const int nb = w >> 2;          // wave's N tile phase (0 or 1)
  const int win = blockIdx.x;
  const float* xw = x + (size_t)win * N_TOK * C_DIM;

  // phase 0: vectorized load of x, dword zero-fill of padded A buffers
  for (int i = tid; i < N_TOK * 48; i += 256) {
    int r = i / 48, c4 = (i - r * 48) * 4;
    *reinterpret_cast<uint4*>(xres + (size_t)r * XS + c4) =
        *reinterpret_cast<const uint4*>(xw + (size_t)r * C_DIM + c4);
  }
  {
    uint* hz = (uint*)hbuf;
    for (int i = tid; i < NPAD * HS / 2; i += 256) hz[i] = 0u;
    uint* pz = (uint*)Pbuf;
    for (int i = tid; i < NPAD * PS / 2; i += 256) pz[i] = 0u;
  }
  __syncthreads();

  // phase 1: LN1
  layernorm_to_bf16(xres, XS, hbuf, HS, ln1_g, ln1_b, w, lane);
  __syncthreads();

  // phase 2: QKV gemm, [64x192]@[192x576]; A tiles hoisted (mi fixed per wave)
  const float qscale = 0.17677669529663688110f;  // 1/sqrt(HEAD_DIM)
  {
    bf16x16 a6[6];
#pragma unroll
    for (int kt = 0; kt < 6; ++kt) a6[kt] = load_tile_a(hbuf, mi16, kt * 32, HS, lane);
    for (int s = 0; s < 18; ++s) {
      const int ni = nb + 2 * s;
      floatx8 acc = {};
#pragma unroll
      for (int kt = 0; kt < 6; ++kt)
        acc = wmma_bf16(a6[kt], load_tile_b(qkv_wT, ni * 16, kt * 32, C_DIM, lane), acc);
      const int c = ni * 16 + n;
      const float bias = qkv_b[c];
      const bool isq = (ni < 12);   // wave-uniform -> scalar branch
      const bool isv = (ni >= 24);  // wave-uniform -> scalar branch
      if (!isv) {
        const float scale = isq ? qscale : 1.f;
#pragma unroll
        for (int i = 0; i < 8; ++i) {
          int r = mi16 + half * 8 + i;
          qkvb[(size_t)r * QS + c] = (__bf16)((acc[i] + bias) * scale);
        }
      } else {
#pragma unroll
        for (int i = 0; i < 8; ++i) {
          int r = mi16 + half * 8 + i;
          float v = acc[i] + bias;
          qkvb[(size_t)r * QS + c] = (__bf16)v;
          vT[(size_t)(c - 2 * C_DIM) * VS + r] = (__bf16)v;  // V transposed
        }
      }
    }
  }
  __syncthreads();

  // phase 3: attention, head by head
  for (int h = 0; h < NHEAD; ++h) {
    // S = q k^T + rel bias (2 tiles per wave, shared A), branchless masking
    {
      bf16x16 aq = load_tile_a(qkvb, mi16, h * 32, QS, lane);
#pragma unroll
      for (int s = 0; s < 2; ++s) {
        const int ni = nb + 2 * s;
        floatx8 zacc = {};
        floatx8 acc =
            wmma_bf16(aq, load_tile_b(qkvb, ni * 16, C_DIM + h * 32, QS, lane), zacc);
        const int col = ni * 16 + n;
        const int colc = (col < N_TOK) ? col : (N_TOK - 1);
        const float* brow = bias_full + (size_t)h * N_TOK * N_TOK + colc;
#pragma unroll
        for (int i = 0; i < 8; ++i) {
          int row = mi16 + half * 8 + i;
          int rowc = (row < N_TOK) ? row : (N_TOK - 1);
          float bv = brow[rowc * N_TOK];
          float sv = (col < N_TOK) ? (acc[i] + bv) : -1e30f;
          Sbuf[(size_t)row * SS + col] = sv;
        }
      }
    }
    __syncthreads();
    // softmax: lane owns cols {2*lane, 2*lane+1}; packed b32 store into P
    for (int r = w; r < N_TOK; r += NWAVE) {
      const float* srow = Sbuf + (size_t)r * SS;
      float a0 = srow[2 * lane];
      float a1 = srow[2 * lane + 1];
      float m = fmaxf(a0, a1);
#pragma unroll
      for (int off = 16; off > 0; off >>= 1) m = fmaxf(m, __shfl_xor(m, off, 32));
      float e0 = __expf(a0 - m), e1 = __expf(a1 - m);
      float s = e0 + e1;
#pragma unroll
      for (int off = 16; off > 0; off >>= 1) s += __shfl_xor(s, off, 32);
      float inv = 1.f / s;
      union { __bf16 b[2]; uint u; } pk;
      pk.b[0] = (__bf16)(e0 * inv);
      pk.b[1] = (__bf16)(e1 * inv);
      *reinterpret_cast<uint*>(Pbuf + (size_t)r * PS + 2 * lane) = pk.u;
    }
    __syncthreads();
    // out_h = P @ V, one 16x16 tile per wave, K = 2x32
    {
      floatx8 acc = {};
#pragma unroll
      for (int kt = 0; kt < 2; ++kt) {
        bf16x16 a = load_tile_a(Pbuf, mi16, kt * 32, PS, lane);
        bf16x16 b = load_tile_b(vT, h * 32 + nb * 16, kt * 32, VS, lane);
        acc = wmma_bf16(a, b, acc);
      }
#pragma unroll
      for (int i = 0; i < 8; ++i) {
        int r = mi16 + half * 8 + i;
        aout[(size_t)r * AS + h * 32 + nb * 16 + n] = (__bf16)acc[i];
      }
    }
    __syncthreads();
  }

  // phase 4: proj + residual -> xres; A tiles hoisted
  {
    bf16x16 ap[6];
#pragma unroll
    for (int kt = 0; kt < 6; ++kt) ap[kt] = load_tile_a(aout, mi16, kt * 32, AS, lane);
#pragma unroll
    for (int s = 0; s < 6; ++s) {
      const int ni = nb + 2 * s;
      floatx8 acc = {};
#pragma unroll
      for (int kt = 0; kt < 6; ++kt)
        acc = wmma_bf16(ap[kt], load_tile_b(proj_wT, ni * 16, kt * 32, C_DIM, lane), acc);
      const int c = ni * 16 + n;
      const float bias = proj_b[c];
#pragma unroll
      for (int i = 0; i < 8; ++i) {
        int r = mi16 + half * 8 + i;
        size_t ix = (size_t)r * XS + c;
        xres[ix] = xres[ix] + acc[i] + bias;
      }
    }
  }
  __syncthreads();

  // phase 5: LN2 -> hbuf
  layernorm_to_bf16(xres, XS, hbuf, HS, ln2_g, ln2_b, w, lane);
  __syncthreads();

  // phase 6: MLP; 768 hidden in 4 chunks of 192; 6 fc2 tiles resident per wave
  floatx8 acc2[6];
  {
    floatx8 zero = {};
#pragma unroll
    for (int j = 0; j < 6; ++j) acc2[j] = zero;
  }

  for (int ch = 0; ch < 4; ++ch) {
    // fc1 chunk + exact-erf GELU -> gbuf
    {
      bf16x16 ah[6];
#pragma unroll
      for (int kt = 0; kt < 6; ++kt) ah[kt] = load_tile_a(hbuf, mi16, kt * 32, HS, lane);
#pragma unroll
      for (int s = 0; s < 6; ++s) {
        const int ni = nb + 2 * s;
        floatx8 acc = {};
#pragma unroll
        for (int kt = 0; kt < 6; ++kt)
          acc = wmma_bf16(ah[kt],
                          load_tile_b(fc1_wT, ch * C_DIM + ni * 16, kt * 32, C_DIM, lane),
                          acc);
        const int cg = ch * C_DIM + ni * 16 + n;
        const float bias = fc1_b[cg];
#pragma unroll
        for (int i = 0; i < 8; ++i) {
          int r = mi16 + half * 8 + i;
          float g = acc[i] + bias;
          float ge = 0.5f * g * (1.f + erff(g * 0.70710678118654752f));
          gbuf[(size_t)r * GS + ni * 16 + n] = (__bf16)ge;
        }
      }
    }
    __syncthreads();
    // fc2 partial accumulate; shared A tiles across the wave's 6 output tiles
    {
      bf16x16 ag[6];
#pragma unroll
      for (int kt = 0; kt < 6; ++kt) ag[kt] = load_tile_a(gbuf, mi16, kt * 32, GS, lane);
#pragma unroll
      for (int j = 0; j < 6; ++j) {
        const int ni = nb + 2 * j;
#pragma unroll
        for (int kt = 0; kt < 6; ++kt)
          acc2[j] = wmma_bf16(
              ag[kt],
              load_tile_b(fc2_wT, ni * 16, ch * C_DIM + kt * 32, 4 * C_DIM, lane),
              acc2[j]);
      }
    }
    __syncthreads();
  }

  // phase 7: final residual into xres (tile-exclusive), then coalesced store
#pragma unroll
  for (int j = 0; j < 6; ++j) {
    const int ni = nb + 2 * j;
    const int c = ni * 16 + n;
    const float bias = fc2_b[c];
#pragma unroll
    for (int i = 0; i < 8; ++i) {
      int r = mi16 + half * 8 + i;
      size_t ix = (size_t)r * XS + c;
      xres[ix] = xres[ix] + acc2[j][i] + bias;
    }
  }
  __syncthreads();
  {
    float* outw = out + (size_t)win * N_TOK * C_DIM;
    for (int i = tid; i < N_TOK * 48; i += 256) {
      int r = i / 48, c4 = (i - r * 48) * 4;
      *reinterpret_cast<uint4*>(outw + (size_t)r * C_DIM + c4) =
          *reinterpret_cast<const uint4*>(xres + (size_t)r * XS + c4);
    }
  }
}

extern "C" void kernel_launch(void* const* d_in, const int* in_sizes, int n_in,
                              void* d_out, int out_size, void* d_ws, size_t ws_size,
                              hipStream_t stream) {
  (void)n_in; (void)out_size; (void)ws_size;
  const float* x      = (const float*)d_in[0];
  const float* ln1_g  = (const float*)d_in[1];
  const float* ln1_b  = (const float*)d_in[2];
  const float* qkv_w  = (const float*)d_in[3];
  const float* qkv_b  = (const float*)d_in[4];
  const float* rbt    = (const float*)d_in[5];
  const float* proj_w = (const float*)d_in[6];
  const float* proj_b = (const float*)d_in[7];
  const float* ln2_g  = (const float*)d_in[8];
  const float* ln2_b  = (const float*)d_in[9];
  const float* fc1_w  = (const float*)d_in[10];
  const float* fc1_b  = (const float*)d_in[11];
  const float* fc2_w  = (const float*)d_in[12];
  const float* fc2_b  = (const float*)d_in[13];
  float* out = (float*)d_out;
  char* ws = (char*)d_ws;

  __bf16* qkv_wT  = (__bf16*)(ws + WS_QKVW);   // [576][192]
  __bf16* proj_wT = (__bf16*)(ws + WS_PROJW);  // [192][192]
  __bf16* fc1_wT  = (__bf16*)(ws + WS_FC1W);   // [768][192]
  __bf16* fc2_wT  = (__bf16*)(ws + WS_FC2W);   // [192][768]
  float*  bias_f  = (float*)(ws + WS_BIAS);    // [6][49][49]

  const int nwin = in_sizes[0] / (N_TOK * C_DIM);

  transpose_to_bf16<<<(192 * 576 + 255) / 256, 256, 0, stream>>>(qkv_w, qkv_wT, 192, 576);
  transpose_to_bf16<<<(192 * 192 + 255) / 256, 256, 0, stream>>>(proj_w, proj_wT, 192, 192);
  transpose_to_bf16<<<(192 * 768 + 255) / 256, 256, 0, stream>>>(fc1_w, fc1_wT, 192, 768);
  transpose_to_bf16<<<(768 * 192 + 255) / 256, 256, 0, stream>>>(fc2_w, fc2_wT, 768, 192);
  build_rel_bias<<<(NHEAD * N_TOK * N_TOK + 255) / 256, 256, 0, stream>>>(rbt, bias_f);

  static_cast<void>(hipFuncSetAttribute(
      reinterpret_cast<const void*>(swin_block_kernel),
      hipFuncAttributeMaxDynamicSharedMemorySize, SMEM_BYTES));

  swin_block_kernel<<<nwin, 256, SMEM_BYTES, stream>>>(
      x, ln1_g, ln1_b, qkv_wT, qkv_b, bias_f, proj_wT, proj_b, ln2_g, ln2_b,
      fc1_wT, fc1_b, fc2_wT, fc2_b, out);
}
